// TTBN_81303730913379
// MI455X (gfx1250) — matrix operand order repaired and checked by reading, not compiled, not run on hardware
//
#include <hip/hip_runtime.h>
#include <hip/hip_bf16.h>

// ---------------------------------------------------------------------------
// Tensor-train contraction chain (194 steps) with per-step global BatchNorm.
// Strategy (MI455X / gfx1250, wave32):
//   * ONE persistent kernel; y-state register-resident for the whole chain.
//   * Each step = V_WMMA_F32_16X16X32_F16 per 16-sample tile:
//       K=32 encodes (c', r, i); A = block-diag weights (rows 8..15 zero),
//       B[k,n] = y[n,c',r]*x[n,i]  (contiguous order k = 16*half + slot in both A and B).
//   * BN stats: wave shuffle-reduction (lanes 16..31 hold zero rows of D),
//     float atomic adds to per-step accumulators, then a device-wide
//     release/acquire split barrier on a per-step counter.
//   * A-operands for all 194 steps are pre-packed into the exact per-lane
//     WMMA register layout by a setup kernel (194 KB, L2-resident).
//   * x streamed with b128 loads, 4 steps per load, one group prefetched.
// ---------------------------------------------------------------------------

typedef __attribute__((ext_vector_type(16))) _Float16 v16h;
typedef __attribute__((ext_vector_type(8)))  float    v8f;
typedef __attribute__((ext_vector_type(4)))  float    v4f;

#define NSAMP     32768
#define LENGTH    196
#define NSTEPS    194          // middle steps with BN
#define ROWBYTES  784          // 4*196 floats per sample row group
#define BLOCKS    64
#define TPB       256
#define NWAVES    (BLOCKS * TPB / 32)   // 512
#define TILES     4            // sample tiles (of 16) per wave: 512*4*16 = 32768
#define INV_M     (1.0f / 65536.0f)     // 1/(N*CH)
#define BN_EPS    1e-5f

// ws layout
#define SUMS_OFF  0                       // float sums[194][8] (sum[h], sumsq[h])
#define CNT_OFF   (NSTEPS * 8 * 4)        // int cnt[194]
#define AW_OFF    7168                    // 256-aligned; _Float16 Aw[194][32][16]

// ---------------------------------------------------------------------------
// Setup: zero per-step accumulators/counters and build per-lane A operands.
// A (16x32 f16) layout: lane L holds row M=L%16; lanes 0-15 slots 0..15 = K0..15,
// lanes 16-31 slots 0..15 = K16..31.
// Wbig[m=(c,h), k=(c',r,i)] = (c==c') ? w_mid[s][r][i][h][c] : 0 ; rows 8..15 = 0.
// ---------------------------------------------------------------------------
__global__ __launch_bounds__(TPB) void tt_setup(const float* __restrict__ w_mid,
                                                float* __restrict__ sums,
                                                int* __restrict__ cnt,
                                                _Float16* __restrict__ Aw) {
  int tid = blockIdx.x * TPB + threadIdx.x;
  if (tid < NSTEPS * 8) sums[tid] = 0.0f;
  if (tid < NSTEPS)     cnt[tid]  = 0;
  if (tid < NSTEPS * 512) {
    int s    = tid >> 9;
    int rem  = tid & 511;
    int lane = rem >> 4;
    int t    = rem & 15;
    int m    = lane & 15;
    int lh   = lane >> 4;
    float v = 0.0f;
    if (m < 8) {
      int c = m >> 2, h = m & 3;
      int k = lh * 16 + t;
      int cp = k >> 4, r = (k >> 2) & 3, i = k & 3;
      if (cp == c) v = w_mid[(((s * 4 + r) * 4 + i) * 4 + h) * 2 + c];
    }
    Aw[tid] = (_Float16)v;
  }
}

// ---------------------------------------------------------------------------
// Persistent main kernel.
// ---------------------------------------------------------------------------
__global__ __launch_bounds__(TPB) void tt_main(const float* __restrict__ x,
                                               const float* __restrict__ wf,
                                               const float* __restrict__ wl,
                                               const float* __restrict__ gmm,
                                               const float* __restrict__ bet,
                                               const _Float16* __restrict__ Aw,
                                               float* __restrict__ sums,
                                               int* __restrict__ cnt,
                                               float* __restrict__ out) {
  const int tid  = blockIdx.x * TPB + threadIdx.x;
  const int wv   = tid >> 5;
  const int lane = threadIdx.x & 31;
  const int col  = lane & 15;        // sample-in-tile = WMMA column / B column
  const int half = lane >> 4;        // 0 -> c'=0 K-rows, 1 -> c'=1 K-rows

  int n[TILES];
  const float* xb[TILES];
#pragma unroll
  for (int t = 0; t < TILES; ++t) {
    n[t]  = (wv * TILES + t) * 16 + col;
    xb[t] = x + (size_t)n[t] * ROWBYTES;
  }

  // x double-buffered in groups of 4 steps (16B-aligned: 196 = 4*49)
  v4f cur[TILES][4], nxt[TILES][4];
#pragma unroll
  for (int t = 0; t < TILES; ++t)
#pragma unroll
    for (int i = 0; i < 4; ++i) {
      cur[t][i] = *(const v4f*)(xb[t] + i * 196);
      nxt[t][i] = *(const v4f*)(xb[t] + i * 196 + 4);
    }

  float g4[4], b4[4];
#pragma unroll
  for (int h = 0; h < 4; ++h) { g4[h] = gmm[h]; b4[h] = bet[h]; }

  // Step 0: y[n, c=half, h] = sum_i x[n,i,0] * w_first[i,h,c]
  float yl[TILES][4];
#pragma unroll
  for (int t = 0; t < TILES; ++t)
#pragma unroll
    for (int h = 0; h < 4; ++h) {
      float acc = 0.0f;
#pragma unroll
      for (int i = 0; i < 4; ++i) {
        float w0 = wf[(i * 4 + h) * 2];      // uniform -> s_load
        float w1 = wf[(i * 4 + h) * 2 + 1];
        acc += cur[t][i][0] * (half ? w1 : w0);
      }
      yl[t][h] = acc;
    }

  const v8f cz = {};

  for (int gq = 0; gq <= 48; ++gq) {
#pragma unroll
    for (int el = 0; el < 4; ++el) {
      int s = gq * 4 + el;
      if (s < 1 || s > NSTEPS) continue;           // steps 1..194

      // A operand for this step: 32B per lane, coalesced, L2-resident
      const _Float16* ap = Aw + (size_t)(s - 1) * 512 + lane * 16;
      v16h a = *(const v16h*)ap;
      if (s < NSTEPS)
        __builtin_prefetch(Aw + (size_t)s * 512 + lane * 16, 0, 1);

      v8f d[TILES];
#pragma unroll
      for (int t = 0; t < TILES; ++t) {
        // B[k = half*16 + r*4 + i, n=col] = y[n, c'=half, r] * x[n, i, s]
        v16h bb = {};
#pragma unroll
        for (int r = 0; r < 4; ++r)
#pragma unroll
          for (int i = 0; i < 4; ++i)
            bb[r * 4 + i] = (_Float16)(yl[t][r] * cur[t][i][el]);
        d[t] = __builtin_amdgcn_wmma_f32_16x16x32_f16(
            false, a, false, bb, (short)0, cz, false, false);
      }

      // BN statistics: rows m=h (c=0) and m=4+h (c=1); lanes 16..31 hold zeros.
      float sh[4], sq[4];
#pragma unroll
      for (int h = 0; h < 4; ++h) {
        float a0 = 0.0f, q0 = 0.0f;
#pragma unroll
        for (int t = 0; t < TILES; ++t) {
          float u = d[t][h], v = d[t][h + 4];
          a0 += u + v;
          q0 += u * u + v * v;
        }
#pragma unroll
        for (int off = 16; off; off >>= 1) {
          a0 += __shfl_xor(a0, off, 32);
          q0 += __shfl_xor(q0, off, 32);
        }
        sh[h] = a0; sq[h] = q0;
      }

      float* ss = sums + (size_t)(s - 1) * 8;
      int*   cc = cnt + (s - 1);
      if (lane == 0) {
#pragma unroll
        for (int h = 0; h < 4; ++h) {
          atomicAdd(ss + h,     sh[h]);
          atomicAdd(ss + 4 + h, sq[h]);
        }
        __hip_atomic_fetch_add(cc, 1, __ATOMIC_RELEASE, __HIP_MEMORY_SCOPE_AGENT);
      }
      // device-wide split barrier: per-step counter, acquire on read
      while (__hip_atomic_load(cc, __ATOMIC_ACQUIRE, __HIP_MEMORY_SCOPE_AGENT) < NWAVES)
        __builtin_amdgcn_s_sleep(1);

      float mean[4], inv[4];
#pragma unroll
      for (int h = 0; h < 4; ++h) {
        float su = __hip_atomic_load(ss + h,     __ATOMIC_RELAXED, __HIP_MEMORY_SCOPE_AGENT);
        float s2 = __hip_atomic_load(ss + 4 + h, __ATOMIC_RELAXED, __HIP_MEMORY_SCOPE_AGENT);
        float mu  = su * INV_M;
        float var = s2 * INV_M - mu * mu;           // biased var (matches jnp.var)
        mean[h] = mu;
        inv[h]  = rsqrtf(var + BN_EPS) * g4[h];
      }

      // Normalize and rebuild per-lane y: half=0 keeps own c=0 rows,
      // half=1 fetches c=1 rows from partner lane (xor 16).
#pragma unroll
      for (int t = 0; t < TILES; ++t) {
        float yn1[4];
#pragma unroll
        for (int h = 0; h < 4; ++h)
          yn1[h] = (d[t][4 + h] - mean[h]) * inv[h] + b4[h];
#pragma unroll
        for (int r = 0; r < 4; ++r) {
          float ours   = (d[t][r] - mean[r]) * inv[r] + b4[r];
          float theirs = __shfl_xor(yn1[r], 16, 32);
          yl[t][r] = half ? theirs : ours;
        }
      }
    }

    // rotate x buffers; prefetch group gq+2
    if (gq < 48) {
#pragma unroll
      for (int t = 0; t < TILES; ++t)
#pragma unroll
        for (int i = 0; i < 4; ++i) {
          cur[t][i] = nxt[t][i];
          if (gq < 47)
            nxt[t][i] = *(const v4f*)(xb[t] + i * 196 + (gq + 2) * 4);
        }
    }
  }

  // Last step (s=195, element 3 of group 48, now in cur):
  // out[n, c=half, o] = sum_{r,i} y[c,r] * x[n,i,195] * w_last[r][i][o][c]
#pragma unroll
  for (int t = 0; t < TILES; ++t) {
    float xv[4];
#pragma unroll
    for (int i = 0; i < 4; ++i) xv[i] = cur[t][i][3];
    float o10[10];
#pragma unroll
    for (int o = 0; o < 10; ++o) {
      float acc = 0.0f;
#pragma unroll
      for (int r = 0; r < 4; ++r)
#pragma unroll
        for (int i = 0; i < 4; ++i) {
          int idx = ((r * 4 + i) * 10 + o) * 2;
          float w0 = wl[idx];          // uniform -> s_load
          float w1 = wl[idx + 1];
          acc += yl[t][r] * xv[i] * (half ? w1 : w0);
        }
      o10[o] = acc;
    }
    float* op = out + (size_t)n[t] * 20 + half * 10;
#pragma unroll
    for (int o = 0; o < 10; ++o) op[o] = o10[o];
  }
}

// ---------------------------------------------------------------------------
extern "C" void kernel_launch(void* const* d_in, const int* in_sizes, int n_in,
                              void* d_out, int out_size, void* d_ws, size_t ws_size,
                              hipStream_t stream) {
  const float* x       = (const float*)d_in[0];
  const float* w_first = (const float*)d_in[1];
  const float* w_mid   = (const float*)d_in[2];
  const float* w_last  = (const float*)d_in[3];
  const float* gamma   = (const float*)d_in[4];
  const float* beta    = (const float*)d_in[5];
  float* out = (float*)d_out;

  char* ws = (char*)d_ws;
  float*     sums = (float*)(ws + SUMS_OFF);
  int*       cnt  = (int*)(ws + CNT_OFF);
  _Float16*  Aw   = (_Float16*)(ws + AW_OFF);   // ~194 KB total ws usage

  // Build WMMA-layout weights + zero per-step barrier state (every launch,
  // so replays are deterministic w.r.t. barrier/accumulator state).
  tt_setup<<<(NSTEPS * 512 + TPB - 1) / TPB, TPB, 0, stream>>>(w_mid, sums, cnt, Aw);

  // Persistent cooperative kernel: 64 blocks (co-resident on MI455X),
  // 512 wave32s, 4 sixteen-sample tiles per wave = full batch.
  tt_main<<<BLOCKS, TPB, 0, stream>>>(x, w_first, w_last, gamma, beta,
                                      Aw, sums, cnt, out);
}
